// DenseGridNet_46677704572931
// MI455X (gfx1250) — matrix operand order repaired and checked by reading, not compiled, not run on hardware
//
#include <hip/hip_runtime.h>

typedef _Float16 f16;
typedef __attribute__((ext_vector_type(16))) _Float16 v16h;
typedef __attribute__((ext_vector_type(8)))  _Float16 v8h;
typedef __attribute__((ext_vector_type(8)))  float    v8f;

#define RES0 512
#define RES1 264
#define RES2 16
#define NBLOCKS 512
#define NTHREADS 256

// Wave-local LDS ordering: all DS ops of this wave complete before continuing.
__device__ __forceinline__ void lds_fence() {
    asm volatile("s_wait_dscnt 0" ::: "memory");
}

__device__ __forceinline__ v16h cat8(v8h lo, v8h hi) {
    return __builtin_shufflevector(lo, hi, 0,1,2,3,4,5,6,7,8,9,10,11,12,13,14,15);
}

// 16-bit A fragment (16x32 tile): lanes 0-15 = rows, lane>>4 selects K-half.
// Per lane: K = [kb, kb+8) in VGPR0-3, K = [kb+16, kb+24) in VGPR4-7, kb = (lane>>4)*8.
__device__ __forceinline__ v16h load_afrag(const f16* base, int lane, int rowBase, int kchunk, int stride) {
    int row = rowBase + (lane & 15);
    int kb  = kchunk + ((lane >> 4) << 3);
    const f16* p = base + row * stride + kb;
    v8h lo = *(const v8h*)(p);
    v8h hi = *(const v8h*)(p + 16);
    return cat8(lo, hi);
}

// 16-bit B fragment (32x16 tile) from transposed weights Wt[n][k]:
// lane n = column, K = [kb, kb+8) then [kb+16, kb+24), kb = kchunk + (lane>>4)*8.
__device__ __forceinline__ v16h load_bfrag(const f16* base, int lane, int ntile, int kchunk, int stride) {
    int n  = (ntile << 4) + (lane & 15);
    int kb = kchunk + ((lane >> 4) << 3);
    const f16* p = base + n * stride + kb;
    v8h lo = *(const v8h*)(p);
    v8h hi = *(const v8h*)(p + 16);
    return cat8(lo, hi);
}

__device__ __forceinline__ v8f wmma16(v16h a, v16h b, v8f c) {
    return __builtin_amdgcn_wmma_f32_16x16x32_f16(false, a, false, b, (short)0, c, false, false);
}

__device__ __forceinline__ void grid_feat(const float* __restrict__ tab, int res,
                                          float u, float v, float* f) {
    float sx = u * (float)res, sy = v * (float)res;
    int   x0 = (int)sx,        y0 = (int)sy;
    float wx = sx - (float)x0, wy = sy - (float)y0;
    int   x1 = min(x0 + 1, res), y1 = min(y0 + 1, res);
    // Reference indexes with stride `res` into a (res+1)^2 table.
    float4 a = *(const float4*)(tab + 4 * (y0 * res + x0));
    float4 b = *(const float4*)(tab + 4 * (y0 * res + x1));
    float4 c = *(const float4*)(tab + 4 * (y1 * res + x0));
    float4 d = *(const float4*)(tab + 4 * (y1 * res + x1));
    float iwx = 1.0f - wx, iwy = 1.0f - wy;
    f[0] = (a.x * iwx + b.x * wx) * iwy + (c.x * iwx + d.x * wx) * wy;
    f[1] = (a.y * iwx + b.y * wx) * iwy + (c.y * iwx + d.y * wx) * wy;
    f[2] = (a.z * iwx + b.z * wx) * iwy + (c.z * iwx + d.z * wx) * wy;
    f[3] = (a.w * iwx + b.w * wx) * iwy + (c.w * iwx + d.w * wx) * wy;
}

__global__ __launch_bounds__(NTHREADS)
void dgn_wmma_kernel(const float* __restrict__ x,
                     const float* __restrict__ emb0,
                     const float* __restrict__ emb1,
                     const float* __restrict__ emb2,
                     const float* __restrict__ w1, const float* __restrict__ b1,
                     const float* __restrict__ w2, const float* __restrict__ b2,
                     const float* __restrict__ w3, const float* __restrict__ b3,
                     float* __restrict__ out, int npts)
{
    // Transposed f16 weight panels (shared by all waves in the block)
    __shared__ __align__(16) f16 s_w1t[64 * 32];   // [n][k], k >= 13 zero
    __shared__ __align__(16) f16 s_w2t[64 * 64];   // [n][k]
    __shared__ __align__(16) f16 s_w3t[16 * 64];   // [n][k], n >= 3 zero
    __shared__ float s_b1[64];
    __shared__ float s_b2[64];
    __shared__ float s_b3[16];
    // Per-wave staging slabs
    __shared__ __align__(16) f16 s_abuf[8][32 * 32];  // embeddings, K padded to 32
    __shared__ __align__(16) f16 s_hbuf[8][32 * 64];  // hidden activations

    const int tid = threadIdx.x;

    // ---- one-time weight prep (f32 -> f16, transpose, pad) ----
    for (int i = tid; i < 64 * 32; i += NTHREADS) {
        int n = i >> 5, k = i & 31;
        s_w1t[i] = (k < 13) ? (f16)w1[k * 64 + n] : (f16)0.0f;
    }
    for (int i = tid; i < 64 * 64; i += NTHREADS) {
        int n = i >> 6, k = i & 63;
        s_w2t[i] = (f16)w2[k * 64 + n];
    }
    for (int i = tid; i < 16 * 64; i += NTHREADS) {
        int n = i >> 6, k = i & 63;
        s_w3t[i] = (n < 3) ? (f16)w3[k * 3 + n] : (f16)0.0f;
    }
    if (tid < 64) { s_b1[tid] = b1[tid]; s_b2[tid] = b2[tid]; }
    if (tid < 16) { s_b3[tid] = (tid < 3) ? b3[tid] : 0.0f; }
    __syncthreads();

    const int lane = tid & 31;
    const int wv   = tid >> 5;
    f16* abuf = s_abuf[wv];
    f16* hbuf = s_hbuf[wv];

    // Loop-invariant B fragments held in VGPRs
    v16h w1f[4], w2f[4][2], w3f[2];
#pragma unroll
    for (int nt = 0; nt < 4; ++nt) {
        w1f[nt]    = load_bfrag(s_w1t, lane, nt, 0, 32);
        w2f[nt][0] = load_bfrag(s_w2t, lane, nt, 0, 64);
        w2f[nt][1] = load_bfrag(s_w2t, lane, nt, 32, 64);
    }
    w3f[0] = load_bfrag(s_w3t, lane, 0, 0, 64);
    w3f[1] = load_bfrag(s_w3t, lane, 0, 32, 64);

    const int ntiles = (npts + 31) >> 5;
    const int gwave  = blockIdx.x * 8 + wv;
    const int nwaves = gridDim.x * 8;
    const int hi     = lane >> 4;

    for (int t = gwave; t < ntiles; t += nwaves) {
        const int base = t << 5;

        // ---- embedding: each lane builds one padded A row (32 f16) ----
        {
            int p = base + lane;
            if (p >= npts) p = npts - 1;
            const float* xp = x + 3 * p;
            float idf = xp[0], u = xp[1], vv = xp[2];
            float f0[4], f1[4], f2[4];
            grid_feat(emb0, RES0, u, vv, f0);
            grid_feat(emb1, RES1, u, vv, f1);
            grid_feat(emb2, RES2, u, vv, f2);
            f16* ar = abuf + lane * 32;
            v8h z = { (f16)0, (f16)0, (f16)0, (f16)0, (f16)0, (f16)0, (f16)0, (f16)0 };
            *(v8h*)(ar + 8)  = z;
            *(v8h*)(ar + 16) = z;
            *(v8h*)(ar + 24) = z;
            ar[0]  = (f16)idf;
            ar[1]  = (f16)f0[0]; ar[2]  = (f16)f0[1]; ar[3]  = (f16)f0[2]; ar[4]  = (f16)f0[3];
            ar[5]  = (f16)f1[0]; ar[6]  = (f16)f1[1]; ar[7]  = (f16)f1[2]; ar[8]  = (f16)f1[3];
            ar[9]  = (f16)f2[0]; ar[10] = (f16)f2[1]; ar[11] = (f16)f2[2]; ar[12] = (f16)f2[3];
        }
        lds_fence();

        // ---- layer 1: h1[32x64] = A[32x32(pad13)] @ W1t, bias+relu -> hbuf ----
#pragma unroll
        for (int m = 0; m < 2; ++m) {
            v16h a = load_afrag(abuf, lane, m << 4, 0, 32);
#pragma unroll
            for (int nt = 0; nt < 4; ++nt) {
                v8f c = {0.f, 0.f, 0.f, 0.f, 0.f, 0.f, 0.f, 0.f};
                c = wmma16(a, w1f[nt], c);
                int   n  = (nt << 4) + (lane & 15);
                float bn = s_b1[n];
#pragma unroll
                for (int r = 0; r < 8; ++r) {
                    int   mm = (m << 4) + (hi << 3) + r;
                    float vr = c[r] + bn;
                    hbuf[mm * 64 + n] = (f16)(vr > 0.0f ? vr : 0.0f);
                }
            }
        }
        lds_fence();

        // ---- layer 2: h2 = h1 @ W2t (K=64), bias+relu, rewrite hbuf in place ----
        v16h a2[2][2];
#pragma unroll
        for (int m = 0; m < 2; ++m) {
            a2[m][0] = load_afrag(hbuf, lane, m << 4, 0, 64);
            a2[m][1] = load_afrag(hbuf, lane, m << 4, 32, 64);
        }
        lds_fence();  // all h1 reads in VGPRs before overwriting
#pragma unroll
        for (int m = 0; m < 2; ++m) {
#pragma unroll
            for (int nt = 0; nt < 4; ++nt) {
                v8f c = {0.f, 0.f, 0.f, 0.f, 0.f, 0.f, 0.f, 0.f};
                c = wmma16(a2[m][0], w2f[nt][0], c);
                c = wmma16(a2[m][1], w2f[nt][1], c);
                int   n  = (nt << 4) + (lane & 15);
                float bn = s_b2[n];
#pragma unroll
                for (int r = 0; r < 8; ++r) {
                    int   mm = (m << 4) + (hi << 3) + r;
                    float vr = c[r] + bn;
                    hbuf[mm * 64 + n] = (f16)(vr > 0.0f ? vr : 0.0f);
                }
            }
        }
        lds_fence();

        // ---- layer 3: out[32x3] = h2 @ W3t (N padded to 16) ----
#pragma unroll
        for (int m = 0; m < 2; ++m) {
            v16h a0 = load_afrag(hbuf, lane, m << 4, 0, 64);
            v16h a1 = load_afrag(hbuf, lane, m << 4, 32, 64);
            v8f c = {0.f, 0.f, 0.f, 0.f, 0.f, 0.f, 0.f, 0.f};
            c = wmma16(a0, w3f[0], c);
            c = wmma16(a1, w3f[1], c);
            int n = lane & 15;
            if (n < 3) {
                float bn = s_b3[n];
#pragma unroll
                for (int r = 0; r < 8; ++r) {
                    int mm = (m << 4) + (hi << 3) + r;
                    int p  = base + mm;
                    if (p < npts) out[p * 3 + n] = c[r] + bn;
                }
            }
        }
    }
}

extern "C" void kernel_launch(void* const* d_in, const int* in_sizes, int n_in,
                              void* d_out, int out_size, void* d_ws, size_t ws_size,
                              hipStream_t stream) {
    const float* x    = (const float*)d_in[0];
    const float* emb0 = (const float*)d_in[1];
    const float* emb1 = (const float*)d_in[2];
    const float* emb2 = (const float*)d_in[3];
    const float* w1   = (const float*)d_in[4];
    const float* b1   = (const float*)d_in[5];
    const float* w2   = (const float*)d_in[6];
    const float* b2   = (const float*)d_in[7];
    const float* w3   = (const float*)d_in[8];
    const float* b3   = (const float*)d_in[9];
    float* out = (float*)d_out;
    int npts = in_sizes[0] / 3;

    dgn_wmma_kernel<<<dim3(NBLOCKS), dim3(NTHREADS), 0, stream>>>(
        x, emb0, emb1, emb2, w1, b1, w2, b2, w3, b3, out, npts);
}
